// LinearAttention_40510131536321
// MI455X (gfx1250) — matrix operand order repaired and checked by reading, not compiled, not run on hardware
//
#include <hip/hip_runtime.h>
#include <hip/hip_bf16.h>

typedef __attribute__((ext_vector_type(16))) _Float16 v16h;
typedef __attribute__((ext_vector_type(8)))  _Float16 v8h;
typedef __attribute__((ext_vector_type(8)))  float    v8f;

#define DIM       512
#define HEADS     8
#define DIM_HEAD  64
#define HIDDEN    512          // HEADS * DIM_HEAD
#define NPIX      4096         // 64*64
#define BATCH     8

__device__ __forceinline__ v16h concat8(v8h lo, v8h hi) {
    return __builtin_shufflevector(lo, hi, 0,1,2,3,4,5,6,7,8,9,10,11,12,13,14,15);
}

__device__ __forceinline__ v8f wmma_f16(v16h a, v16h b, v8f c) {
    // D = A(16x32 f16) * B(32x16 f16) + C(16x16 f32)
    return __builtin_amdgcn_wmma_f32_16x16x32_f16(false, a, false, b, (short)0, c, false, false);
}

// ---------------------------------------------------------------------------
// fp32 -> fp16 conversion (weights)
// ---------------------------------------------------------------------------
__global__ void cvt_f32_to_f16(const float* __restrict__ src,
                               _Float16* __restrict__ dst, int n) {
    int i = blockIdx.x * 256 + threadIdx.x;
    if (i < n) dst[i] = (_Float16)src[i];
}

// ---------------------------------------------------------------------------
// Tiled WMMA GEMM:  C[M,N] = A[M,K] * B[K,N]   (A f16 row-major, shared over z;
// B has per-batch stride; B elements TB converted to f16 on the LDS stage).
// Block tile 128x128, BK=32, 256 threads = 8 waves in 2(M)x4(N), each wave
// owns a 64x32 patch = 4x2 wmma tiles.
// WRITE_F16: store f16 (qkv).  else: store f32 + bias (final y).
// ---------------------------------------------------------------------------
template <typename TB, bool WRITE_F16>
__global__ __launch_bounds__(256)
void gemm_wmma_kernel(const _Float16* __restrict__ A,
                      const TB* __restrict__ B,
                      void* __restrict__ C,
                      const float* __restrict__ bias,
                      int M, int N, int K,
                      long long strideB, long long strideC)
{
    constexpr int LSTR = 40;   // f16 stride: 80 B -> 16B aligned rows, bank-spread
    __shared__ __attribute__((aligned(16))) _Float16 As[128 * LSTR];
    __shared__ __attribute__((aligned(16))) _Float16 Bs[128 * LSTR];  // transposed [n][k]

    const int tid   = threadIdx.x;
    const int wave  = tid >> 5;
    const int lane  = tid & 31;
    const int waveM = wave >> 2;        // 0..1
    const int waveN = wave & 3;         // 0..3
    const int blockM = blockIdx.y * 128;
    const int blockN = blockIdx.x * 128;

    const TB* Bb = B + (long long)blockIdx.z * strideB;

    v8f acc[4][2] = {};

    const int kbA   = (lane & 16) ? 16 : 0;   // A frag: K runs {kbA..kbA+15}
    const int kbB   = (lane & 16) ? 16 : 0;   // B frag: K runs {kbB..kbB+15}
    const int lrow  = lane & 15;

    for (int k0 = 0; k0 < K; k0 += 32) {
        __syncthreads();
        // ---- stage A tile 128x32 (f16 copy) ----
        {
            int row  = tid >> 1;
            int colh = (tid & 1) * 16;
            const _Float16* src = A + (long long)(blockM + row) * K + k0 + colh;
            v8h a0 = *(const v8h*)(src);
            v8h a1 = *(const v8h*)(src + 8);
            *(v8h*)(&As[row * LSTR + colh])     = a0;
            *(v8h*)(&As[row * LSTR + colh + 8]) = a1;
        }
        // ---- stage B tile 32x128, transposed into [n][k], converted to f16 ----
        {
            int krow   = tid >> 3;           // 0..31
            int nstart = (tid & 7) * 16;     // 0..112
            const TB* src = Bb + (long long)(k0 + krow) * N + blockN + nstart;
            #pragma unroll
            for (int i = 0; i < 16; ++i)
                Bs[(nstart + i) * LSTR + krow] = (_Float16)(float)src[i];
        }
        // prefetch next K-step tiles into cache (global_prefetch_b8)
        if (k0 + 32 < K) {
            __builtin_prefetch(A  + (long long)(blockM + (tid >> 1)) * K + k0 + 32, 0, 0);
            __builtin_prefetch(Bb + (long long)(k0 + 32 + (tid >> 3)) * N + blockN + (tid & 7) * 16, 0, 0);
        }
        __syncthreads();

        // ---- build fragments and multiply ----
        v16h afrag[4], bfrag[2];
        #pragma unroll
        for (int mt = 0; mt < 4; ++mt) {
            int row = waveM * 64 + mt * 16 + lrow;
            v8h lo = *(const v8h*)(&As[row * LSTR + kbA]);
            v8h hi = *(const v8h*)(&As[row * LSTR + kbA + 8]);
            afrag[mt] = concat8(lo, hi);
        }
        #pragma unroll
        for (int nt = 0; nt < 2; ++nt) {
            int ncol = waveN * 32 + nt * 16 + lrow;
            v8h lo = *(const v8h*)(&Bs[ncol * LSTR + kbB]);
            v8h hi = *(const v8h*)(&Bs[ncol * LSTR + kbB + 8]);
            bfrag[nt] = concat8(lo, hi);
        }
        #pragma unroll
        for (int mt = 0; mt < 4; ++mt)
            #pragma unroll
            for (int nt = 0; nt < 2; ++nt)
                acc[mt][nt] = wmma_f16(afrag[mt], bfrag[nt], acc[mt][nt]);
    }

    // ---- epilogue:  D layout: VGPR r, lane L -> (M = r + 8*(L>=16), N = L%16)
    const int rbase = (lane & 16) ? 8 : 0;
    #pragma unroll
    for (int mt = 0; mt < 4; ++mt) {
        #pragma unroll
        for (int nt = 0; nt < 2; ++nt) {
            int col = blockN + waveN * 32 + nt * 16 + lrow;
            #pragma unroll
            for (int r = 0; r < 8; ++r) {
                int row = blockM + waveM * 64 + mt * 16 + rbase + r;
                float v = acc[mt][nt][r];
                if (WRITE_F16) {
                    _Float16* dst = (_Float16*)C + (long long)blockIdx.z * strideC;
                    dst[(long long)row * N + col] = (_Float16)v;
                } else {
                    float* dst = (float*)C + (long long)blockIdx.z * strideC;
                    dst[(long long)row * N + col] = v + bias[row];
                }
            }
        }
    }
}

// ---------------------------------------------------------------------------
// Row softmax over the K part of qkv (f16 in/out, f32 math).
// One 256-thread block per row of length 4096.
// ---------------------------------------------------------------------------
__global__ __launch_bounds__(256)
void softmax_rows_kernel(_Float16* __restrict__ qkv)
{
    int rowIdx = blockIdx.x;                 // 0 .. 8*512-1
    int b = rowIdx >> 9;
    int r = rowIdx & 511;
    _Float16* row = qkv + ((long long)b * 1536 + 512 + r) * NPIX;

    __shared__ float red[256];
    int tid = threadIdx.x;
    float vals[16];
    float lmax = -1e30f;
    #pragma unroll
    for (int i = 0; i < 16; ++i) {
        float v = (float)row[tid + i * 256];
        vals[i] = v;
        lmax = fmaxf(lmax, v);
    }
    red[tid] = lmax; __syncthreads();
    for (int s = 128; s > 0; s >>= 1) {
        if (tid < s) red[tid] = fmaxf(red[tid], red[tid + s]);
        __syncthreads();
    }
    float m = red[0]; __syncthreads();

    float lsum = 0.f;
    #pragma unroll
    for (int i = 0; i < 16; ++i) { vals[i] = __expf(vals[i] - m); lsum += vals[i]; }
    red[tid] = lsum; __syncthreads();
    for (int s = 128; s > 0; s >>= 1) {
        if (tid < s) red[tid] += red[tid + s];
        __syncthreads();
    }
    float inv = 1.0f / red[0];
    #pragma unroll
    for (int i = 0; i < 16; ++i)
        row[tid + i * 256] = (_Float16)(vals[i] * inv);
}

// ---------------------------------------------------------------------------
// context[d][e] = sum_n k_sm[d][n] * v[e][n]    per (b,h):  64x64, K=4096.
// A = k rows (contiguous K), B = v rows (contiguous K) -> direct global frags.
// Output stored TRANSPOSED as ct[e][d] (f16) so the next GEMM's A is row-major.
// One 256-thread block per (b,h); wave w: mt = w/2, nt in {2*(w&1), 2*(w&1)+1}.
// ---------------------------------------------------------------------------
__global__ __launch_bounds__(256)
void context_kernel(const _Float16* __restrict__ qkv, _Float16* __restrict__ ct)
{
    int bh = blockIdx.x;                     // 0..63
    int b = bh >> 3, h = bh & 7;
    const _Float16* Kmat = qkv + ((long long)b * 1536 +  512 + h * 64) * NPIX;
    const _Float16* Vmat = qkv + ((long long)b * 1536 + 1024 + h * 64) * NPIX;

    int tid = threadIdx.x, wave = tid >> 5, lane = tid & 31;
    int mt = wave >> 1;
    int ntb = (wave & 1) * 2;
    const int kbA = (lane & 16) ? 16 : 0;
    const int kbB = (lane & 16) ? 16 : 0;
    const int lrow = lane & 15;

    v8f acc[2] = {};
    int arow = mt * 16 + lrow;

    for (int k0 = 0; k0 < NPIX; k0 += 32) {
        const _Float16* ap = Kmat + (long long)arow * NPIX + k0;
        v8h alo = *(const v8h*)(ap + kbA);
        v8h ahi = *(const v8h*)(ap + kbA + 8);
        v16h af = concat8(alo, ahi);
        #pragma unroll
        for (int nt = 0; nt < 2; ++nt) {
            int erow = (ntb + nt) * 16 + lrow;
            const _Float16* bp = Vmat + (long long)erow * NPIX + k0 + kbB;
            v8h blo = *(const v8h*)(bp);
            v8h bhi = *(const v8h*)(bp + 8);
            acc[nt] = wmma_f16(af, concat8(blo, bhi), acc[nt]);
        }
    }

    _Float16* out = ct + (long long)bh * 64 * 64;   // [e][d]
    const int rbase = (lane & 16) ? 8 : 0;
    #pragma unroll
    for (int nt = 0; nt < 2; ++nt) {
        int e = (ntb + nt) * 16 + lrow;
        #pragma unroll
        for (int r = 0; r < 8; ++r) {
            int d = mt * 16 + rbase + r;
            out[e * 64 + d] = (_Float16)acc[nt][r];
        }
    }
}

// ---------------------------------------------------------------------------
// out[e][n] = sum_d ct[e][d] * q[d][n]    per (b,h): M=64, K=64, N=4096.
// Q tile (64 x 128) staged transposed in LDS; ct fragments direct from global.
// Block: (b,h) = blockIdx.y, n-chunk = blockIdx.x*128; wave w owns nt=w, mt=0..3.
// ---------------------------------------------------------------------------
__global__ __launch_bounds__(256)
void apply_kernel(const _Float16* __restrict__ qkv,
                  const _Float16* __restrict__ ct,
                  _Float16* __restrict__ outh)
{
    constexpr int QSTR = 72;   // 144 B rows: 16B aligned, bank-spread
    __shared__ __attribute__((aligned(16))) _Float16 Qs[128 * QSTR];  // [n][d]

    int bh = blockIdx.y;
    int b = bh >> 3, h = bh & 7;
    int n0 = blockIdx.x * 128;
    const _Float16* Qmat = qkv + ((long long)b * 1536 + h * 64) * NPIX;
    const _Float16* Amat = ct + (long long)bh * 64 * 64;   // [e][d]

    int tid = threadIdx.x, wave = tid >> 5, lane = tid & 31;

    // stage Q transposed
    {
        int d = tid >> 2;                 // 0..63
        int nstart = (tid & 3) * 32;      // 0..96
        const _Float16* src = Qmat + (long long)d * NPIX + n0 + nstart;
        #pragma unroll
        for (int i = 0; i < 32; ++i)
            Qs[(nstart + i) * QSTR + d] = src[i];
    }
    __syncthreads();

    const int kbA = (lane & 16) ? 16 : 0;
    const int kbB = (lane & 16) ? 16 : 0;
    const int lrow = lane & 15;
    int nt = wave;                        // 0..7

    v8f acc[4] = {};
    #pragma unroll
    for (int k0 = 0; k0 < 64; k0 += 32) {
        int n = nt * 16 + lrow;
        v8h blo = *(const v8h*)(&Qs[n * QSTR + k0 + kbB]);
        v8h bhi = *(const v8h*)(&Qs[n * QSTR + k0 + kbB + 8]);
        v16h bf = concat8(blo, bhi);
        #pragma unroll
        for (int mt = 0; mt < 4; ++mt) {
            int e = mt * 16 + lrow;
            const _Float16* ap = Amat + (long long)e * 64 + k0;
            v8h alo = *(const v8h*)(ap + kbA);
            v8h ahi = *(const v8h*)(ap + kbA + 8);
            acc[mt] = wmma_f16(concat8(alo, ahi), bf, acc[mt]);
        }
    }

    _Float16* dst = outh + ((long long)b * HIDDEN + h * 64) * NPIX + n0;
    const int rbase = (lane & 16) ? 8 : 0;
    #pragma unroll
    for (int mt = 0; mt < 4; ++mt) {
        int col = nt * 16 + lrow;
        #pragma unroll
        for (int r = 0; r < 8; ++r) {
            int e = mt * 16 + rbase + r;
            dst[(long long)e * NPIX + col] = (_Float16)acc[mt][r];
        }
    }
}

// ---------------------------------------------------------------------------
// Host-side pipeline
// ---------------------------------------------------------------------------
extern "C" void kernel_launch(void* const* d_in, const int* in_sizes, int n_in,
                              void* d_out, int out_size, void* d_ws, size_t ws_size,
                              hipStream_t stream) {
    const float* x      = (const float*)d_in[0];   // [8, 512, 64, 64]
    const float* w_qkv  = (const float*)d_in[1];   // [1536, 512]
    const float* w_out  = (const float*)d_in[2];   // [512, 512]
    const float* b_out  = (const float*)d_in[3];   // [512]
    float*       y      = (float*)d_out;           // [8, 512, 64, 64]

    // workspace layout (all f16, all 16B aligned)
    _Float16* w_qkv_h = (_Float16*)d_ws;                       // 1536*512
    _Float16* w_out_h = w_qkv_h + 1536 * 512;                  // 512*512
    _Float16* qkv_h   = w_out_h + 512 * 512;                   // 8*1536*4096
    _Float16* ct_h    = qkv_h + (long long)BATCH * 1536 * NPIX;// 8*8*64*64
    _Float16* out_h   = ct_h + (long long)BATCH * HEADS * 64 * 64; // 8*512*4096

    // 0) convert weights to f16
    cvt_f32_to_f16<<<(1536 * 512 + 255) / 256, 256, 0, stream>>>(w_qkv, w_qkv_h, 1536 * 512);
    cvt_f32_to_f16<<<(512  * 512 + 255) / 256, 256, 0, stream>>>(w_out, w_out_h, 512 * 512);

    // 1) qkv = w_qkv @ x   per batch: [1536,512]x[512,4096] -> f16
    gemm_wmma_kernel<float, true><<<dim3(NPIX / 128, 1536 / 128, BATCH), 256, 0, stream>>>(
        w_qkv_h, x, qkv_h, nullptr, 1536, NPIX, DIM,
        (long long)DIM * NPIX, (long long)1536 * NPIX);

    // 2) softmax over k rows (8*512 rows of 4096)
    softmax_rows_kernel<<<BATCH * 512, 256, 0, stream>>>(qkv_h);

    // 3) context = k_sm @ v^T per (b,h), stored transposed [e][d]
    context_kernel<<<BATCH * HEADS, 256, 0, stream>>>(qkv_h, ct_h);

    // 4) out = ct @ q per (b,h)
    apply_kernel<<<dim3(NPIX / 128, BATCH * HEADS), 256, 0, stream>>>(qkv_h, ct_h, out_h);

    // 5) y = w_out @ out + b_out  per batch: [512,512]x[512,4096] -> f32
    gemm_wmma_kernel<_Float16, false><<<dim3(NPIX / 128, HIDDEN / 128, BATCH), 256, 0, stream>>>(
        w_out_h, out_h, y, b_out, HIDDEN, NPIX, HIDDEN,
        (long long)HIDDEN * NPIX, (long long)HIDDEN * NPIX);
}
